// BrainPathwayAnalysis_63058709840308
// MI455X (gfx1250) — matrix operand order repaired and checked by reading, not compiled, not run on hardware
//
#include <hip/hip_runtime.h>
#include <math.h>

#define EPSLN 1e-5f
#define B_  32
#define P_  2048
#define I_  4096
#define L_  256

// vrf = sqrt(2048)/2 ; INV_VRF = 1/vrf
#define INV_VRF 0.04419417382415922f
#define SOFTC   0.5f

typedef __attribute__((ext_vector_type(16))) _Float16 v16h;
typedef __attribute__((ext_vector_type(8)))  _Float16 v8h;
typedef __attribute__((ext_vector_type(8)))  float    v8f;

union V16 { v16h v; v8h h[2]; };

__device__ __forceinline__ float lrelu(float x) { return x >= 0.f ? x : 0.2f * x; }

// ---------------------------------------------------------------------------
// Stage A1: qm[b,p,:] = lrelu( lrelu(LN(pathway[b,p]*qw, q_g,q_b)) @ qw2 )  -> f16
// one 64-thread block per (b,p) row
// ---------------------------------------------------------------------------
__global__ void __launch_bounds__(64) qm_kernel(
    const float* __restrict__ pathway, const float* __restrict__ qw,
    const float* __restrict__ qw2, const float* __restrict__ qg,
    const float* __restrict__ qb, _Float16* __restrict__ qm)
{
    const int row = blockIdx.x;        // b*P + p
    const int t   = threadIdx.x;       // 0..63
    const float s = pathway[row];
    const float x = s * qw[t];

    __shared__ float xs[64];
    __shared__ float red[4];

    float sum = x, sq = x * x;
    #pragma unroll
    for (int o = 16; o > 0; o >>= 1) {
        sum += __shfl_xor(sum, o, 32);
        sq  += __shfl_xor(sq,  o, 32);
    }
    const int w = t >> 5;
    if ((t & 31) == 0) { red[w] = sum; red[2 + w] = sq; }
    __syncthreads();
    const float m = (red[0] + red[1]) * (1.0f / 64.0f);
    const float v = (red[2] + red[3]) * (1.0f / 64.0f) - m * m;
    float y = (x - m) * rsqrtf(v + EPSLN) * qg[t] + qb[t];
    y = lrelu(y);
    xs[t] = y;
    __syncthreads();

    float acc = 0.f;
    #pragma unroll 8
    for (int k = 0; k < 64; ++k) acc = fmaf(xs[k], qw2[k * 64 + t], acc);
    acc = lrelu(acc);
    qm[(size_t)row * 64 + t] = (_Float16)acc;
}

// ---------------------------------------------------------------------------
// Stage A2: km[b,i,:] = LN(img_t @ kw)  -> f16
//           vm[b,i]   = lrelu(LN(img_t @ vw)) @ vw2  -> f32
// one 64-thread block per (b,i) row
// ---------------------------------------------------------------------------
__global__ void __launch_bounds__(64) kmvm_kernel(
    const float* __restrict__ img, const float* __restrict__ kw,
    const float* __restrict__ vw, const float* __restrict__ vw2,
    const float* __restrict__ kg, const float* __restrict__ kb,
    const float* __restrict__ vg, const float* __restrict__ vb,
    _Float16* __restrict__ km, float* __restrict__ vm)
{
    const int row = blockIdx.x;         // b*I + i
    const int b   = row >> 12;          // /4096
    const int i   = row & (I_ - 1);
    const int t   = threadIdx.x;

    const float c0 = img[(size_t)(b * 4 + 0) * I_ + i];
    const float c1 = img[(size_t)(b * 4 + 1) * I_ + i];
    const float c2 = img[(size_t)(b * 4 + 2) * I_ + i];
    const float c3 = img[(size_t)(b * 4 + 3) * I_ + i];

    const float xk = c0 * kw[t] + c1 * kw[64 + t] + c2 * kw[128 + t] + c3 * kw[192 + t];
    const float xv = c0 * vw[t] + c1 * vw[64 + t] + c2 * vw[128 + t] + c3 * vw[192 + t];

    __shared__ float red[8];
    float s0 = xk, s1 = xk * xk, s2 = xv, s3 = xv * xv;
    #pragma unroll
    for (int o = 16; o > 0; o >>= 1) {
        s0 += __shfl_xor(s0, o, 32); s1 += __shfl_xor(s1, o, 32);
        s2 += __shfl_xor(s2, o, 32); s3 += __shfl_xor(s3, o, 32);
    }
    const int w = t >> 5;
    if ((t & 31) == 0) { red[w] = s0; red[2 + w] = s1; red[4 + w] = s2; red[6 + w] = s3; }
    __syncthreads();
    const float mk = (red[0] + red[1]) * (1.0f / 64.0f);
    const float vk = (red[2] + red[3]) * (1.0f / 64.0f) - mk * mk;
    const float mv = (red[4] + red[5]) * (1.0f / 64.0f);
    const float vv = (red[6] + red[7]) * (1.0f / 64.0f) - mv * mv;

    const float yk = (xk - mk) * rsqrtf(vk + EPSLN) * kg[t] + kb[t];
    km[(size_t)row * 64 + t] = (_Float16)yk;

    float yv = (xv - mv) * rsqrtf(vv + EPSLN) * vg[t] + vb[t];
    yv = lrelu(yv);
    float pv = yv * vw2[t];
    #pragma unroll
    for (int o = 16; o > 0; o >>= 1) pv += __shfl_xor(pv, o, 32);
    __syncthreads();                     // red reuse
    if ((t & 31) == 0) red[w] = pv;
    __syncthreads();
    if (t == 0) vm[row] = red[0] + red[1];
}

// ---------------------------------------------------------------------------
// zero helper
// ---------------------------------------------------------------------------
__global__ void zero_kernel(float* __restrict__ p, int n)
{
    int idx = blockIdx.x * blockDim.x + threadIdx.x;
    if (idx < n) p[idx] = 0.f;
}

// ---------------------------------------------------------------------------
// Stage B: attn[b,p,i] = softsign(relu(qm[b,p,:]·km[b,i,:]) / vrf)
//          out[b,p]   += attn[b,p,i] * vm[b,i]
// One wave per (b, p-tile, i-chunk). 8 waves / 256-thread block.
// K=64 via two v_wmma_f32_16x16x32_f16.
// Tasks: 32 batches * 128 p-tiles * 8 chunks (of 32 i-tiles) = 32768 waves.
// ---------------------------------------------------------------------------
__global__ void __launch_bounds__(256, 2) attn_kernel(
    const _Float16* __restrict__ qm, const _Float16* __restrict__ km,
    const float* __restrict__ vm, float* __restrict__ attn_out,
    float* __restrict__ out_acc)
{
    const int wave  = blockIdx.x * 8 + (threadIdx.x >> 5);
    const int lane  = threadIdx.x & 31;
    const int chunk = wave & 7;
    const int ptile = (wave >> 3) & 127;
    const int b     = wave >> 10;

    const int p0   = ptile * 16;
    const int m16  = lane & 15;          // column N for B/D, row M%16 for A
    const int ksel = lane >> 4;

    // ---- A fragments: qm rows p0..p0+15, K=0..63 (ISA 16-bit A layout) ----
    const _Float16* arow = qm + ((size_t)b * P_ + p0 + m16) * 64;
    V16 a_lo, a_hi;
    a_lo.h[0] = *(const v8h*)(arow +      ksel * 8);   // K 0..7 / 8..15
    a_lo.h[1] = *(const v8h*)(arow + 16 + ksel * 8);   // K 16..23 / 24..31
    a_hi.h[0] = *(const v8h*)(arow + 32 + ksel * 8);   // K 32..39 / 40..47
    a_hi.h[1] = *(const v8h*)(arow + 48 + ksel * 8);   // K 48..55 / 56..63

    float outacc[8];
    #pragma unroll
    for (int r = 0; r < 8; ++r) outacc[r] = 0.f;

    const _Float16* kmb = km + (size_t)b * I_ * 64;
    const float*    vmb = vm + (size_t)b * I_;
    const size_t attn_base = (size_t)b * P_ * I_;
    const int it0 = chunk * 32;

    for (int it = it0; it < it0 + 32; ++it) {
        const int i0 = it * 16;
        // B fragment: column n = m16 (km row i0+n), lanes 0-15 hold K 0..7,16..23,
        // lanes 16-31 hold K 8..15,24..31 (low frag) / +32 (high frag)
        const _Float16* brow = kmb + (size_t)(i0 + m16) * 64 + ksel * 8;
        __builtin_prefetch(brow + 16 * 64, 0, 3);      // next i-tile
        V16 b_lo, b_hi;
        b_lo.h[0] = *(const v8h*)(brow);
        b_lo.h[1] = *(const v8h*)(brow + 16);
        b_hi.h[0] = *(const v8h*)(brow + 32);
        b_hi.h[1] = *(const v8h*)(brow + 48);

        const float vmv = vmb[i0 + m16];

        v8f acc = {0.f, 0.f, 0.f, 0.f, 0.f, 0.f, 0.f, 0.f};
        acc = __builtin_amdgcn_wmma_f32_16x16x32_f16(
                  false, a_lo.v, false, b_lo.v, (short)0, acc, false, false);
        acc = __builtin_amdgcn_wmma_f32_16x16x32_f16(
                  false, a_hi.v, false, b_hi.v, (short)0, acc, false, false);

        // D layout: lane holds column n=m16, rows (ksel*8 + r)
        float* dst = attn_out + attn_base + (size_t)(p0 + ksel * 8) * I_ + (i0 + m16);
        #pragma unroll
        for (int r = 0; r < 8; ++r) {
            float s = fmaxf(acc[r], 0.f) * INV_VRF;      // relu + /vrf
            // softsign: s >= 0, denom in [0.5, inf) -> hw rcp is safe (~1 ulp)
            s = s * __builtin_amdgcn_rcpf(s + SOFTC);
            __builtin_nontemporal_store(s, dst + (size_t)r * I_);
            outacc[r] = fmaf(s, vmv, outacc[r]);
        }
    }

    // reduce out over the 16 columns held by each half-wave; 1 atomic per row
    #pragma unroll
    for (int r = 0; r < 8; ++r) {
        float v = outacc[r];
        #pragma unroll
        for (int o = 1; o < 16; o <<= 1) v += __shfl_xor(v, o, 32);
        if (m16 == 0)
            atomicAdd(out_acc + b * P_ + p0 + ksel * 8 + r, v);
    }
}

// ---------------------------------------------------------------------------
// Stage C: classifier head. One 256-thread block per batch.
// h = lrelu(LN(out, n1)); z = h @ w1.T + b1; h2 = lrelu(LN(z, n2));
// logit = h2 @ w2.T + b2
// ---------------------------------------------------------------------------
__global__ void __launch_bounds__(256) head_kernel(
    const float* __restrict__ outv,
    const float* __restrict__ n1g, const float* __restrict__ n1b,
    const float* __restrict__ w1,  const float* __restrict__ b1,
    const float* __restrict__ n2g, const float* __restrict__ n2b,
    const float* __restrict__ w2,  const float* __restrict__ b2,
    float* __restrict__ logits)
{
    const int b = blockIdx.x;
    const int t = threadIdx.x;
    __shared__ float h1[P_];
    __shared__ float red[16];

    const float* orow = outv + (size_t)b * P_;
    float loc[8];
    float sum = 0.f, sq = 0.f;
    #pragma unroll
    for (int j = 0; j < 8; ++j) {
        float x = orow[t + j * 256];
        loc[j] = x; sum += x; sq += x * x;
    }
    #pragma unroll
    for (int o = 16; o > 0; o >>= 1) {
        sum += __shfl_xor(sum, o, 32);
        sq  += __shfl_xor(sq,  o, 32);
    }
    const int w = t >> 5;
    if ((t & 31) == 0) { red[w] = sum; red[8 + w] = sq; }
    __syncthreads();
    sum = 0.f; sq = 0.f;
    #pragma unroll
    for (int k = 0; k < 8; ++k) { sum += red[k]; sq += red[8 + k]; }
    const float m1 = sum * (1.0f / P_);
    const float v1 = sq * (1.0f / P_) - m1 * m1;
    const float rs1 = rsqrtf(v1 + EPSLN);
    #pragma unroll
    for (int j = 0; j < 8; ++j) {
        int p = t + j * 256;
        h1[p] = lrelu((loc[j] - m1) * rs1 * n1g[p] + n1b[p]);
    }
    __syncthreads();   // h1 ready; also guards red reuse

    float z = b1[t];
    const float* w1row = w1 + (size_t)t * P_;
    #pragma unroll 8
    for (int p = 0; p < P_; ++p) z = fmaf(h1[p], w1row[p], z);

    // LN over L=256
    float s2 = z, q2 = z * z;
    #pragma unroll
    for (int o = 16; o > 0; o >>= 1) {
        s2 += __shfl_xor(s2, o, 32);
        q2 += __shfl_xor(q2, o, 32);
    }
    if ((t & 31) == 0) { red[w] = s2; red[8 + w] = q2; }
    __syncthreads();
    s2 = 0.f; q2 = 0.f;
    #pragma unroll
    for (int k = 0; k < 8; ++k) { s2 += red[k]; q2 += red[8 + k]; }
    const float m2 = s2 * (1.0f / L_);
    const float v2 = q2 * (1.0f / L_) - m2 * m2;
    float h2 = lrelu((z - m2) * rsqrtf(v2 + EPSLN) * n2g[t] + n2b[t]);

    float part = h2 * w2[t];
    #pragma unroll
    for (int o = 16; o > 0; o >>= 1) part += __shfl_xor(part, o, 32);
    __syncthreads();                     // red reuse
    if ((t & 31) == 0) red[w] = part;
    __syncthreads();
    if (t == 0) {
        float tot = 0.f;
        #pragma unroll
        for (int k = 0; k < 8; ++k) tot += red[k];
        logits[b] = tot + b2[0];
    }
}

// ---------------------------------------------------------------------------
extern "C" void kernel_launch(void* const* d_in, const int* in_sizes, int n_in,
                              void* d_out, int out_size, void* d_ws, size_t ws_size,
                              hipStream_t stream)
{
    const float* img     = (const float*)d_in[0];
    const float* pathway = (const float*)d_in[1];
    const float* qw      = (const float*)d_in[2];
    const float* qw2     = (const float*)d_in[3];
    const float* kw      = (const float*)d_in[4];
    const float* vw      = (const float*)d_in[5];
    const float* vw2     = (const float*)d_in[6];
    const float* q_g     = (const float*)d_in[7];
    const float* q_b     = (const float*)d_in[8];
    const float* k_g     = (const float*)d_in[9];
    const float* k_b     = (const float*)d_in[10];
    const float* v_g     = (const float*)d_in[11];
    const float* v_b     = (const float*)d_in[12];
    const float* n1_g    = (const float*)d_in[13];
    const float* n1_b    = (const float*)d_in[14];
    const float* n2_g    = (const float*)d_in[15];
    const float* n2_b    = (const float*)d_in[16];
    const float* w1      = (const float*)d_in[17];
    const float* b1      = (const float*)d_in[18];
    const float* w2      = (const float*)d_in[19];
    const float* b2      = (const float*)d_in[20];

    // d_out = [logits (B), attn (B*P*I), out (B*P)]
    float* logits = (float*)d_out;
    float* attn   = logits + B_;
    float* outv   = attn + (size_t)B_ * P_ * I_;

    // workspace: qm f16 (8MB) | km f16 (16MB) | vm f32 (0.5MB)
    _Float16* qm = (_Float16*)d_ws;
    _Float16* km = qm + (size_t)B_ * P_ * 64;
    float*    vm = (float*)(km + (size_t)B_ * I_ * 64);

    qm_kernel<<<B_ * P_, 64, 0, stream>>>(pathway, qw, qw2, q_g, q_b, qm);
    kmvm_kernel<<<B_ * I_, 64, 0, stream>>>(img, kw, vw, vw2, k_g, k_b, v_g, v_b, km, vm);
    zero_kernel<<<(B_ * P_ + 255) / 256, 256, 0, stream>>>(outv, B_ * P_);
    zero_kernel<<<1, 32, 0, stream>>>(logits, B_);
    attn_kernel<<<(B_ * 128 * 8) / 8, 256, 0, stream>>>(qm, km, vm, attn, outv);
    head_kernel<<<B_, 256, 0, stream>>>(outv, n1_g, n1_b, w1, b1, n2_g, n2_b, w2, b2, logits);
}
